// ViTDiscriminator_55748675502562
// MI455X (gfx1250) — matrix operand order repaired and checked by reading, not compiled, not run on hardware
//
#include <hip/hip_runtime.h>
#include <cstddef>

// ---------------- model constants ----------------
#define BATCH 16
#define TOK   1025          // T = 1024 patches + cls
#define DIM   384
#define NHEAD 4
#define HDIM  96
#define NBLK  3
#define FFDIM 1536
#define NCLS  2
#define KPATCH 768          // CIN*P*P
#define MPATCH 16384        // B * 1024
#define MTOK  (BATCH*TOK)   // 16400
#define ATT_SCALE 0.10206207261596575f  // 1/sqrt(96)

// ---------------- types ----------------
typedef __attribute__((ext_vector_type(16))) __bf16          v16bf;
typedef __attribute__((ext_vector_type(8)))  float           v8f;
typedef __attribute__((ext_vector_type(8)))  unsigned short  u16x8;
typedef int v4i __attribute__((vector_size(16)));   // matches builtin param type

union ABFrag { v16bf bf; u16x8 h[2]; };

__device__ inline unsigned short f2bf(float f) {
  unsigned int u = __builtin_bit_cast(unsigned int, f);
  unsigned int r = (u + 0x7FFFu + ((u >> 16) & 1u)) >> 16;   // RNE
  return (unsigned short)r;
}
__device__ inline float bf2f(unsigned short h) {
  unsigned int u = ((unsigned int)h) << 16;
  return __builtin_bit_cast(float, u);
}

__device__ inline v8f wmma_bf16(const ABFrag& a, const ABFrag& b, v8f c) {
  return __builtin_amdgcn_wmma_f32_16x16x32_bf16(false, a.bf, false, b.bf,
                                                 (short)0, c, false, false);
}

// ---------------- async copy-to-LDS (gfx1250 path, guarded) ----------------
#if defined(__gfx1250__) && \
    __has_builtin(__builtin_amdgcn_global_load_async_to_lds_b128) && \
    __has_builtin(__builtin_amdgcn_s_wait_asynccnt)
#define HAVE_ASYNC 1
#else
#define HAVE_ASYNC 0
#endif

__device__ inline void copy16(const unsigned short* g, unsigned short* l) {
#if HAVE_ASYNC
  __builtin_amdgcn_global_load_async_to_lds_b128(
      (__attribute__((address_space(1))) v4i*)g,
      (__attribute__((address_space(3))) v4i*)l, 0, 0);
#else
  *(u16x8*)l = *(const u16x8*)g;
#endif
}
__device__ inline void async_wait() {
#if HAVE_ASYNC
  __builtin_amdgcn_s_wait_asynccnt(0);
#endif
}

// ---------------- epilogue ids ----------------
#define EPI_PATCH 0   // h[b,1+p,:] = acc + conv_b + pos_emb[1+p]
#define EPI_QKV   1   // qkv[((b*4+h)*T+t)*96+hd] = bf16(acc)
#define EPI_RESID 2   // h[m,:] += acc + bias
#define EPI_RELU  3   // out_bf16[m*N+n] = bf16(relu(acc+bias))

// =====================================================================
// bf16 WMMA GEMM: C(M,N) = A(M,K) * Wt^T + epilogue
// A: bf16 row-major [M][K].  Wt: bf16 [N][K] (pre-transposed weights).
// 256 thr = 8 waves; block tile 128(M)x128(N), BK=32, ping-pong LDS;
// wave tile 32x64 -> 8 wmma per K-step.
// =====================================================================
template<int EPI>
__global__ __launch_bounds__(256) void gemm_wmma(
    const unsigned short* __restrict__ A, const unsigned short* __restrict__ Wt,
    int M, int K, int N,
    const float* __restrict__ bias, const float* __restrict__ extra,
    float* __restrict__ outF, unsigned short* __restrict__ outB)
{
  __shared__ __attribute__((aligned(16))) unsigned short As[2][128][32];
  __shared__ __attribute__((aligned(16))) unsigned short Bs[2][128][32];

  const int tid  = threadIdx.x;
  const int lane = tid & 31;
  const int wid  = tid >> 5;
  const int waveM = wid & 3;            // 4 waves along M (32 rows each)
  const int waveN = wid >> 2;           // 2 waves along N (64 cols each)
  const int m0 = blockIdx.x * 128;
  const int n0 = blockIdx.y * 128;

  v8f acc[2][4];
#pragma unroll
  for (int i = 0; i < 2; ++i)
#pragma unroll
    for (int j = 0; j < 4; ++j)
      acc[i][j] = (v8f){0.f,0.f,0.f,0.f,0.f,0.f,0.f,0.f};

  const int r    = lane & 15;
  const int aoff = (lane < 16) ? 0 : 8;    // A K-sub-base (ISA 16-bit A layout)
  const int boff = (lane < 16) ? 0 : 8;    // B K-sub-base

  auto stageA = [&](int buf, int k0) {
#pragma unroll
    for (int i = 0; i < 2; ++i) {
      int c = tid + 256 * i;              // 512 chunks of 8 (128 rows x 4)
      int row = c >> 2, kc = (c & 3) << 3;
      int gm = m0 + row;
      unsigned short* dst = &As[buf][row][kc];
      if (gm < M) {
        copy16(A + (size_t)gm * K + (k0 + kc), dst);
      } else {
        u16x8 z = {0,0,0,0,0,0,0,0};
        *(u16x8*)dst = z;
      }
    }
  };
  auto stageB = [&](int buf, int k0) {
#pragma unroll
    for (int i = 0; i < 2; ++i) {
      int c = tid + 256 * i;              // 512 chunks of 8 (128 n-rows x 4)
      int n = c >> 2, kc = (c & 3) << 3;
      copy16(Wt + (size_t)(n0 + n) * K + (k0 + kc), &Bs[buf][n][kc]);
    }
  };

  stageA(0, 0);
  stageB(0, 0);
  async_wait();
  __syncthreads();

  const int nsteps = K >> 5;
  for (int s = 0; s < nsteps; ++s) {
    const int cur = s & 1, nxt = cur ^ 1;
    if (s + 1 < nsteps) { stageA(nxt, (s + 1) << 5); stageB(nxt, (s + 1) << 5); }

    ABFrag af[2], bfr[4];
#pragma unroll
    for (int mi = 0; mi < 2; ++mi) {
      const unsigned short* p = &As[cur][waveM * 32 + mi * 16 + r][aoff];
      af[mi].h[0] = *(const u16x8*)p;
      af[mi].h[1] = *(const u16x8*)(p + 16);
    }
#pragma unroll
    for (int ni = 0; ni < 4; ++ni) {
      const unsigned short* p = &Bs[cur][waveN * 64 + ni * 16 + r][boff];
      bfr[ni].h[0] = *(const u16x8*)p;
      bfr[ni].h[1] = *(const u16x8*)(p + 16);
    }
#pragma unroll
    for (int mi = 0; mi < 2; ++mi)
#pragma unroll
      for (int ni = 0; ni < 4; ++ni)
        acc[mi][ni] = wmma_bf16(af[mi], bfr[ni], acc[mi][ni]);

    async_wait();
    __syncthreads();
  }

  // ---- epilogue: C layout lane<16 -> rows j, lane>=16 -> rows 8+j; col = lane&15
  const int cl = lane & 15, rOff = (lane < 16) ? 0 : 8;
#pragma unroll
  for (int mi = 0; mi < 2; ++mi)
#pragma unroll
    for (int ni = 0; ni < 4; ++ni) {
#pragma unroll
      for (int jj = 0; jj < 8; ++jj) {
        int m = m0 + waveM * 32 + mi * 16 + rOff + jj;
        int n = n0 + waveN * 64 + ni * 16 + cl;
        if (m >= M) continue;
        float a = acc[mi][ni][jj];
        if (EPI == EPI_PATCH) {
          int b = m >> 10, p = m & 1023;
          size_t row = (size_t)b * TOK + 1 + p;
          outF[row * DIM + n] = a + bias[n] + extra[(size_t)(1 + p) * DIM + n];
        } else if (EPI == EPI_QKV) {
          int b = m / TOK, t = m % TOK;
          int hh = n / HDIM, hd = n % HDIM;
          outB[(((size_t)b * NHEAD + hh) * TOK + t) * HDIM + hd] = f2bf(a);
        } else if (EPI == EPI_RESID) {
          size_t o = (size_t)m * DIM + n;
          outF[o] += a + bias[n];
        } else { // EPI_RELU
          float v = a + bias[n];
          if (v < 0.f) v = 0.f;
          outB[(size_t)m * N + n] = f2bf(v);
        }
      }
    }
}

// =====================================================================
// Flash-style distance attention, one (b,head,128-query-tile) per block.
// score = sqrt(max(|q|^2+|k|^2-2qk, 0)) / sqrt(HD), online softmax, P*V.
// =====================================================================
__global__ __launch_bounds__(256) void flash_attn(
    const unsigned short* __restrict__ Q, const unsigned short* __restrict__ Kb,
    const unsigned short* __restrict__ Vb, unsigned short* __restrict__ Ob)
{
  __shared__ __attribute__((aligned(16))) unsigned short Qs[128][96];
  __shared__ __attribute__((aligned(16))) unsigned short Ks[32][96];
  __shared__ __attribute__((aligned(16))) unsigned short Vt[96][32];   // transposed
  __shared__ float q2s[128];
  __shared__ float k2s[32];
  __shared__ __attribute__((aligned(16))) float          Ss[8][16][32];
  __shared__ __attribute__((aligned(16))) unsigned short Ps[8][16][32];
  __shared__ float rowm[8][16], rowl[8][16], rowa[8][16];

  const int tid = threadIdx.x, lane = tid & 31, w = tid >> 5;
  const int bh = blockIdx.x;              // b*NHEAD + h
  const int q0 = blockIdx.y * 128;

  // ---- stage Q tile (async) ----
#pragma unroll
  for (int i = 0; i < 6; ++i) {
    int ch = tid + 256 * i;               // 1536 chunks of 8
    int row = ch / 12, cc = (ch % 12) * 8;
    int t = q0 + row;
    unsigned short* dst = &Qs[row][cc];
    if (t < TOK) {
      copy16(Q + ((size_t)bh * TOK + t) * HDIM + cc, dst);
    } else {
      u16x8 z = {0,0,0,0,0,0,0,0};
      *(u16x8*)dst = z;
    }
  }
  async_wait();
  __syncthreads();
  if (tid < 128) {
    float s = 0.f;
    for (int c = 0; c < HDIM; ++c) { float v = bf2f(Qs[tid][c]); s += v * v; }
    q2s[tid] = s;
  }
  if (tid < 128) { rowm[tid >> 4][tid & 15] = -3e38f; rowl[tid >> 4][tid & 15] = 0.f; }

  v8f acc[6];
#pragma unroll
  for (int nc = 0; nc < 6; ++nc) acc[nc] = (v8f){0.f,0.f,0.f,0.f,0.f,0.f,0.f,0.f};

  const int r    = lane & 15;
  const int aoff = (lane < 16) ? 0 : 8;
  const int boff = (lane < 16) ? 0 : 8;
  const int rOff = (lane < 16) ? 0 : 8;

  for (int j = 0; j < 33; ++j) {          // ceil(1025/32) key tiles
    __syncthreads();
    // ---- stage K tile (row-major, async) ----
#pragma unroll
    for (int i = 0; i < 2; ++i) {
      int ch = tid + 256 * i;             // 384 chunks
      if (ch < 384) {
        int row = ch / 12, cc = (ch % 12) * 8;
        int t = j * 32 + row;
        unsigned short* dst = &Ks[row][cc];
        if (t < TOK) {
          copy16(Kb + ((size_t)bh * TOK + t) * HDIM + cc, dst);
        } else {
          u16x8 z = {0,0,0,0,0,0,0,0};
          *(u16x8*)dst = z;
        }
      }
    }
    // ---- stage V tile transposed: Vt[hd][key] ----
#pragma unroll
    for (int i = 0; i < 12; ++i) {
      int idx = tid + 256 * i;            // 3072 elements
      int key = idx / 96, hd = idx % 96;
      int t = j * 32 + key;
      Vt[hd][key] = (t < TOK) ? Vb[((size_t)bh * TOK + t) * HDIM + hd] : (unsigned short)0;
    }
    async_wait();
    __syncthreads();
    if (tid < 32) {
      float s = 0.f;
      for (int c = 0; c < HDIM; ++c) { float v = bf2f(Ks[tid][c]); s += v * v; }
      k2s[tid] = s;
    }
    __syncthreads();

    // ---- S = Q * K^T  (16x32 per wave, 3 wmma k-steps x 2 key halves) ----
    v8f s2[2];
    s2[0] = (v8f){0.f,0.f,0.f,0.f,0.f,0.f,0.f,0.f};
    s2[1] = (v8f){0.f,0.f,0.f,0.f,0.f,0.f,0.f,0.f};
#pragma unroll
    for (int kc = 0; kc < 96; kc += 32) {
      ABFrag a;
      const unsigned short* ap = &Qs[w * 16 + r][kc + aoff];
      a.h[0] = *(const u16x8*)ap;
      a.h[1] = *(const u16x8*)(ap + 16);
#pragma unroll
      for (int nh = 0; nh < 2; ++nh) {
        ABFrag b;
        const unsigned short* bp = &Ks[nh * 16 + r][kc + boff];
        b.h[0] = *(const u16x8*)bp;
        b.h[1] = *(const u16x8*)(bp + 16);
        s2[nh] = wmma_bf16(a, b, s2[nh]);
      }
    }
#pragma unroll
    for (int nh = 0; nh < 2; ++nh)
#pragma unroll
      for (int jj = 0; jj < 8; ++jj)
        Ss[w][rOff + jj][nh * 16 + (lane & 15)] = s2[nh][jj];

    // ---- online softmax (lane&15 = row, lane>>4 = column half) ----
    const int ch = (lane >> 4) * 16;
    float qq = q2s[w * 16 + r];
    float pmax = -3e38f;
#pragma unroll
    for (int c = 0; c < 16; ++c) {
      int col = ch + c, kg = j * 32 + col;
      float sv = qq + k2s[col] - 2.f * Ss[w][r][col];
      sv = sqrtf(fmaxf(sv, 0.f)) * ATT_SCALE;
      if (kg >= TOK) sv = -3e38f;
      Ss[w][r][col] = sv;
      pmax = fmaxf(pmax, sv);
    }
    pmax = fmaxf(pmax, __shfl_xor(pmax, 16, 32));
    float mold = rowm[w][r];
    float mnew = fmaxf(mold, pmax);
    float alpha = __expf(mold - mnew);
    float psum = 0.f;
#pragma unroll
    for (int c = 0; c < 16; ++c) {
      float p = __expf(Ss[w][r][ch + c] - mnew);
      Ps[w][r][ch + c] = f2bf(p);
      psum += p;
    }
    psum += __shfl_xor(psum, 16, 32);
    if (lane < 16) {
      rowl[w][r] = rowl[w][r] * alpha + psum;
      rowm[w][r] = mnew;
      rowa[w][r] = alpha;
    }

    // ---- rescale accumulators, then acc += P * V ----
    ABFrag pf;
    const unsigned short* pp = &Ps[w][r][aoff];
    pf.h[0] = *(const u16x8*)pp;
    pf.h[1] = *(const u16x8*)(pp + 16);
#pragma unroll
    for (int nc = 0; nc < 6; ++nc) {
      ABFrag vf;
      const unsigned short* vp = &Vt[nc * 16 + r][boff];
      vf.h[0] = *(const u16x8*)vp;
      vf.h[1] = *(const u16x8*)(vp + 16);
#pragma unroll
      for (int jj = 0; jj < 8; ++jj) acc[nc][jj] *= rowa[w][rOff + jj];
      acc[nc] = wmma_bf16(pf, vf, acc[nc]);
    }
  }

  // ---- write O = acc / l, scattered back to (b, t, h*96+hd) bf16 ----
  const int b = bh >> 2, hh = bh & 3;
#pragma unroll
  for (int nc = 0; nc < 6; ++nc) {
#pragma unroll
    for (int jj = 0; jj < 8; ++jj) {
      int row = rOff + jj;
      int t = q0 + w * 16 + row;
      if (t < TOK) {
        float o = acc[nc][jj] / rowl[w][row];
        int hd = nc * 16 + (lane & 15);
        Ob[((size_t)b * TOK + t) * DIM + hh * HDIM + hd] = f2bf(o);
      }
    }
  }
}

// =====================================================================
// Weight prep: f32 -> bf16, transposed to [N][K] where needed
// =====================================================================
__global__ void wt_copy_k(const float* __restrict__ src, unsigned short* __restrict__ dst) {
  int idx = blockIdx.x * 256 + threadIdx.x;
  dst[idx] = f2bf(src[idx]);
}
__global__ void wt_transpose_k(const float* __restrict__ src, unsigned short* __restrict__ dst,
                               int K, int N) {
  int idx = blockIdx.x * 256 + threadIdx.x;   // over N*K
  int n = idx / K, k = idx % K;
  dst[idx] = f2bf(src[(size_t)k * N + n]);
}

// =====================================================================
// Small helper kernels
// =====================================================================
__global__ void im2col_k(const float* __restrict__ x, unsigned short* __restrict__ Ap) {
  size_t idx = (size_t)blockIdx.x * 256 + threadIdx.x;   // MPATCH*KPATCH
  int m = (int)(idx / KPATCH), k = (int)(idx % KPATCH);
  int b = m >> 10, pp = m & 1023, ph = pp >> 5, pw = pp & 31;
  int c = k >> 8, rr = k & 255, py = rr >> 4, px = rr & 15;
  float v = x[(((size_t)b * 3 + c) * 512 + ph * 16 + py) * 512 + pw * 16 + px];
  Ap[idx] = f2bf(v);
}

__global__ void cls_init_k(const float* __restrict__ cls, const float* __restrict__ pos,
                           float* __restrict__ h) {
  int b = blockIdx.x, d = threadIdx.x;
  h[(size_t)b * TOK * DIM + d] = cls[d] + pos[d];
}

__global__ void layernorm_k(const float* __restrict__ h, const float* __restrict__ s,
                            const float* __restrict__ bb, unsigned short* __restrict__ out) {
  int row = blockIdx.x, tid = threadIdx.x;          // 128 threads
  const float* x = h + (size_t)row * DIM;
  float v0 = x[tid], v1 = x[tid + 128], v2 = x[tid + 256];
  __shared__ float red[128];
  red[tid] = v0 + v1 + v2;
  __syncthreads();
  for (int o = 64; o > 0; o >>= 1) { if (tid < o) red[tid] += red[tid + o]; __syncthreads(); }
  float mean = red[0] / DIM;
  __syncthreads();
  float d0 = v0 - mean, d1 = v1 - mean, d2 = v2 - mean;
  red[tid] = d0 * d0 + d1 * d1 + d2 * d2;
  __syncthreads();
  for (int o = 64; o > 0; o >>= 1) { if (tid < o) red[tid] += red[tid + o]; __syncthreads(); }
  float rstd = rsqrtf(red[0] / DIM + 1e-5f);
  unsigned short* o = out + (size_t)row * DIM;
  o[tid]       = f2bf(d0 * rstd * s[tid]       + bb[tid]);
  o[tid + 128] = f2bf(d1 * rstd * s[tid + 128] + bb[tid + 128]);
  o[tid + 256] = f2bf(d2 * rstd * s[tid + 256] + bb[tid + 256]);
}

__global__ void head_k(const unsigned short* __restrict__ hn, const float* __restrict__ pw,
                       const float* __restrict__ pb, const int* __restrict__ tgt,
                       float* __restrict__ out) {
  __shared__ float sl[32];
  __shared__ float sp[16];
  int tid = threadIdx.x;                             // 64 threads
  if (tid < 32) {
    int b = tid >> 1, c = tid & 1;
    const unsigned short* row = hn + (size_t)b * TOK * DIM;   // cls token row
    float a = pb[c];
    for (int d = 0; d < DIM; ++d) a += bf2f(row[d]) * pw[d * NCLS + c];
    sl[tid] = a;
    out[b * NCLS + c] = a;
  }
  __syncthreads();
  if (tid < 16) {
    float l0 = sl[tid * 2], l1 = sl[tid * 2 + 1];
    float m = fmaxf(l0, l1);
    float lse = m + __logf(__expf(l0 - m) + __expf(l1 - m));
    sp[tid] = (tgt[tid] ? l1 : l0) - lse;
  }
  __syncthreads();
  if (tid == 0) {
    float s = 0.f;
    for (int i = 0; i < 16; ++i) s += sp[i];
    out[BATCH * NCLS] = -s / BATCH;
  }
}

// =====================================================================
// host launcher
// =====================================================================
extern "C" void kernel_launch(void* const* d_in, const int* in_sizes, int n_in,
                              void* d_out, int out_size, void* d_ws, size_t ws_size,
                              hipStream_t stream) {
  (void)in_sizes; (void)n_in; (void)out_size; (void)ws_size;
  const float* x      = (const float*)d_in[0];
  const int*   tgt    = (const int*)d_in[1];
  const float* conv_w = (const float*)d_in[2];
  const float* conv_b = (const float*)d_in[3];
  const float* cls    = (const float*)d_in[4];
  const float* pos    = (const float*)d_in[5];
  const float* ln1_s  = (const float*)d_in[6];
  const float* ln1_b  = (const float*)d_in[7];
  const float* ln2_s  = (const float*)d_in[8];
  const float* ln2_b  = (const float*)d_in[9];
  const float* wq     = (const float*)d_in[10];
  const float* wk     = (const float*)d_in[11];
  const float* wv     = (const float*)d_in[12];
  const float* wo     = (const float*)d_in[13];
  const float* bo     = (const float*)d_in[14];
  const float* w1     = (const float*)d_in[15];
  const float* b1     = (const float*)d_in[16];
  const float* w2     = (const float*)d_in[17];
  const float* b2     = (const float*)d_in[18];
  const float* lnf_s  = (const float*)d_in[19];
  const float* lnf_b  = (const float*)d_in[20];
  const float* proj_w = (const float*)d_in[21];
  const float* proj_b = (const float*)d_in[22];
  float* out = (float*)d_out;

  // workspace carve (256B aligned)
  char* cur = (char*)d_ws;
  auto carve = [&](size_t bytes) { void* p = cur; cur += (bytes + 255) & ~(size_t)255; return p; };
  float*          h    = (float*)         carve((size_t)MTOK * DIM * 4);
  unsigned short* hn   = (unsigned short*)carve((size_t)MTOK * DIM * 2);
  unsigned short* Ap   = (unsigned short*)carve((size_t)MPATCH * KPATCH * 2);
  unsigned short* qb   = (unsigned short*)carve((size_t)MTOK * DIM * 2);
  unsigned short* kb   = (unsigned short*)carve((size_t)MTOK * DIM * 2);
  unsigned short* vb   = (unsigned short*)carve((size_t)MTOK * DIM * 2);
  unsigned short* attn = (unsigned short*)carve((size_t)MTOK * DIM * 2);
  unsigned short* ffn  = (unsigned short*)carve((size_t)MTOK * FFDIM * 2);
  // bf16 weights, [N][K] layout
  unsigned short* WcB  = (unsigned short*)carve((size_t)DIM * KPATCH * 2);
  unsigned short* WqB  = (unsigned short*)carve((size_t)NBLK * DIM * DIM * 2);
  unsigned short* WkB  = (unsigned short*)carve((size_t)NBLK * DIM * DIM * 2);
  unsigned short* WvB  = (unsigned short*)carve((size_t)NBLK * DIM * DIM * 2);
  unsigned short* WoB  = (unsigned short*)carve((size_t)NBLK * DIM * DIM * 2);
  unsigned short* W1B  = (unsigned short*)carve((size_t)NBLK * DIM * FFDIM * 2);
  unsigned short* W2B  = (unsigned short*)carve((size_t)NBLK * FFDIM * DIM * 2);

  // ---- weight prep (one-time per launch; lives in L2 thereafter) ----
  wt_copy_k<<<(DIM * KPATCH) / 256, 256, 0, stream>>>(conv_w, WcB);   // already [N][K]
  for (int i = 0; i < NBLK; ++i) {
    const size_t qo = (size_t)i * DIM * DIM, fo = (size_t)i * DIM * FFDIM;
    wt_transpose_k<<<(DIM * DIM) / 256, 256, 0, stream>>>(wq + qo, WqB + qo, DIM, DIM);
    wt_transpose_k<<<(DIM * DIM) / 256, 256, 0, stream>>>(wk + qo, WkB + qo, DIM, DIM);
    wt_transpose_k<<<(DIM * DIM) / 256, 256, 0, stream>>>(wv + qo, WvB + qo, DIM, DIM);
    wt_transpose_k<<<(DIM * DIM) / 256, 256, 0, stream>>>(wo + qo, WoB + qo, DIM, DIM);
    wt_transpose_k<<<(DIM * FFDIM) / 256, 256, 0, stream>>>(w1 + fo, W1B + fo, DIM, FFDIM);
    wt_transpose_k<<<(FFDIM * DIM) / 256, 256, 0, stream>>>(w2 + fo, W2B + fo, FFDIM, DIM);
  }

  const int mtTok = (MTOK + 127) / 128;     // 129

  // 1) patch embedding
  im2col_k<<<(MPATCH * KPATCH) / 256, 256, 0, stream>>>(x, Ap);
  cls_init_k<<<BATCH, DIM, 0, stream>>>(cls, pos, h);
  gemm_wmma<EPI_PATCH><<<dim3(MPATCH / 128, DIM / 128), 256, 0, stream>>>(
      Ap, WcB, MPATCH, KPATCH, DIM, conv_b, pos, h, nullptr);

  // 2) transformer blocks
  for (int i = 0; i < NBLK; ++i) {
    const size_t qo = (size_t)i * DIM * DIM, fo = (size_t)i * DIM * FFDIM;
    layernorm_k<<<MTOK, 128, 0, stream>>>(h, ln1_s + i * DIM, ln1_b + i * DIM, hn);
    gemm_wmma<EPI_QKV><<<dim3(mtTok, DIM / 128), 256, 0, stream>>>(
        hn, WqB + qo, MTOK, DIM, DIM, nullptr, nullptr, nullptr, qb);
    gemm_wmma<EPI_QKV><<<dim3(mtTok, DIM / 128), 256, 0, stream>>>(
        hn, WkB + qo, MTOK, DIM, DIM, nullptr, nullptr, nullptr, kb);
    gemm_wmma<EPI_QKV><<<dim3(mtTok, DIM / 128), 256, 0, stream>>>(
        hn, WvB + qo, MTOK, DIM, DIM, nullptr, nullptr, nullptr, vb);
    flash_attn<<<dim3(BATCH * NHEAD, (TOK + 127) / 128), 256, 0, stream>>>(qb, kb, vb, attn);
    gemm_wmma<EPI_RESID><<<dim3(mtTok, DIM / 128), 256, 0, stream>>>(
        attn, WoB + qo, MTOK, DIM, DIM, bo + i * DIM, nullptr, h, nullptr);
    layernorm_k<<<MTOK, 128, 0, stream>>>(h, ln2_s + i * DIM, ln2_b + i * DIM, hn);
    gemm_wmma<EPI_RELU><<<dim3(mtTok, FFDIM / 128), 256, 0, stream>>>(
        hn, W1B + fo, MTOK, DIM, FFDIM, b1 + i * FFDIM, nullptr, nullptr, ffn);
    gemm_wmma<EPI_RESID><<<dim3(mtTok, DIM / 128), 256, 0, stream>>>(
        ffn, W2B + fo, MTOK, FFDIM, DIM, b2 + i * DIM, nullptr, h, nullptr);
  }

  // 3) final LN + head + loss
  layernorm_k<<<MTOK, 128, 0, stream>>>(h, lnf_s, lnf_b, hn);
  head_k<<<1, 64, 0, stream>>>(hn, proj_w, proj_b, tgt, out);
}
